// BiLSTM_CRF_14044543058062
// MI455X (gfx1250) — compile-verified, matrix operand order checked by reading, not run on hardware
//
#include <hip/hip_runtime.h>
#include <hip/hip_bf16.h>

typedef __attribute__((ext_vector_type(16))) _Float16 v16h;
typedef __attribute__((ext_vector_type(8)))  _Float16 v8h;
typedef __attribute__((ext_vector_type(8)))  float    v8f;
typedef __attribute__((ext_vector_type(4))) unsigned  v4u;
typedef __attribute__((ext_vector_type(8))) int       v8i;
typedef __attribute__((ext_vector_type(4))) int       v4i;

#define V_   32000
#define L_   32
#define E_   512
#define H_   512
#define B_   128
#define T_   512
#define G4H  (4*H_)      // 2048
#define BT   (B_*T_)     // 65536
#define BOS_TAG 1
#define EOS_TAG 2

#if __has_builtin(__builtin_amdgcn_tensor_load_to_lds) && \
    __has_builtin(__builtin_amdgcn_s_wait_tensorcnt)
#define HAVE_TDM 1
#else
#define HAVE_TDM 0
#endif

// ---------------------------------------------------------------------------
// WMMA fragment loader: 16x32 f16 tile from a row-major [rows, ld] matrix.
// ISA layout (CDNA5 ISA, 16-bit A 16x32): lane<16 -> row=lane, K 0..7 & 16..23;
// lane>=16 -> row=lane-16, K 8..15 & 24..31.  Same layout serves B fragments
// sourced from a row-major [N, K] weight (B[k][n] = W[n][k]).
// ---------------------------------------------------------------------------
__device__ inline v16h ld_frag16x32(const _Float16* __restrict__ p0, int ld, int lane) {
  int m  = lane & 15;
  int kb = (lane >> 4) << 3;
  const _Float16* p = p0 + (long)m * ld + kb;
  v8h lo = *(const v8h*)(p);
  v8h hi = *(const v8h*)(p + 16);
  v16h r;
#pragma unroll
  for (int i = 0; i < 8; ++i) { r[i] = lo[i]; r[i + 8] = hi[i]; }
  return r;
}

__device__ inline float sigf(float x) { return 1.0f / (1.0f + __expf(-x)); }

// ---------------------------------------------------------------------------
// Tensor Data Mover: DMA a 2D f16 tile (tile_rows x tile_cols) from a
// row-major global matrix (row stride = `stride` elements) into LDS at
// byte-offset lds_addr.  Descriptor packing per CDNA5 ISA.
// Issue from ONE wave; EXEC is ignored by tensor ops.
// This toolchain exposes the 6-arg builtin:
//   (uint32x4 g0, int32x8 g1, int32x4 g2, int32x4 g3, int32x8 pad, i32 cpol)
// ---------------------------------------------------------------------------
__device__ inline void tdm_load_2d_f16(unsigned lds_addr, const void* gptr,
                                       unsigned tile_cols, unsigned tile_rows,
                                       unsigned stride_elems) {
#if HAVE_TDM
  unsigned long long ga = (unsigned long long)gptr;
  v4u g0;
  g0[0] = 1u;                                   // count=1, user descriptor
  g0[1] = lds_addr;                             // lds_addr  (bits 63:32)
  g0[2] = (unsigned)(ga & 0xffffffffu);         // global_addr[31:0]   (bits 95:64)
  g0[3] = (unsigned)((ga >> 32) & 0x1ffffffu)   // global_addr[56:32]  (bits 120:96)
          | (2u << 30);                         // type=2 ("image")
  v8i g1;
  g1[0] = (int)(1u << 16);                      // data_size=1 -> 2 bytes/elem
  g1[1] = (int)((tile_cols & 0xffffu) << 16);   // tensor_dim0[15:0]  @ bits 63:48
  g1[2] = (int)(((tile_cols >> 16) & 0xffffu)   // tensor_dim0[31:16]
          | ((tile_rows & 0xffffu) << 16));     // tensor_dim1[15:0]
  g1[3] = (int)(((tile_rows >> 16) & 0xffffu)   // tensor_dim1[31:16]
          | ((tile_cols & 0xffffu) << 16));     // tile_dim0
  g1[4] = (int)(tile_rows & 0xffffu);           // tile_dim1 (tile_dim2=0)
  g1[5] = (int)stride_elems;                    // tensor_dim0_stride[31:0]
  g1[6] = 0;                                    // stride0[47:32] | stride1 lo
  g1[7] = 0;
  v4i z4 = {0, 0, 0, 0};
  v8i z8 = {0, 0, 0, 0, 0, 0, 0, 0};
  __builtin_amdgcn_tensor_load_to_lds(g0, g1, z4, z4, z8, 0);
#else
  (void)lds_addr; (void)gptr; (void)tile_cols; (void)tile_rows; (void)stride_elems;
#endif
}

__device__ inline void tdm_wait() {
#if HAVE_TDM
  __builtin_amdgcn_s_wait_tensorcnt(0);
#endif
}

// generic->LDS byte address: hardware truncates generic LDS addresses to [31:0]
__device__ inline unsigned lds_byte_addr(const void* p) {
  return (unsigned)(uintptr_t)p;
}

// ---------------------------------------------------------------------------
// 0) utility kernels
// ---------------------------------------------------------------------------
__global__ void zero_u32_kernel(unsigned* p, int n) {
  int i = blockIdx.x * blockDim.x + threadIdx.x;
  if (i < n) p[i] = 0u;
}

__global__ void cvt_f32_f16_kernel(const float* __restrict__ src,
                                   _Float16* __restrict__ dst, long n) {
  long i = (long)blockIdx.x * blockDim.x + threadIdx.x;
  if (i < n) dst[i] = (_Float16)src[i];
}

__global__ void embed_kernel(const int* __restrict__ x,
                             const float* __restrict__ emb,
                             _Float16* __restrict__ xe) {
  long i = (long)blockIdx.x * blockDim.x + threadIdx.x;
  if (i >= (long)BT * E_) return;
  long tok = i >> 9;
  int  e   = (int)(i & (E_ - 1));
  xe[i] = (_Float16)emb[(long)x[tok] * E_ + e];
}

// ---------------------------------------------------------------------------
// 1) Pregate GEMM: P_d = xe @ W_ih_d^T + b_d    [BT, 4H] f16
//    M=65536 N=2048 K=512.  Block = 8 waves; wave = 16(M) x 64(N).
//    Block's 64 W-rows (64x512 f16 = 64KB) staged in LDS once via TDM;
//    A fragments from global, software-pipelined across the K loop.
// ---------------------------------------------------------------------------
__global__ void pregate_gemm_kernel(const _Float16* __restrict__ xe,
                                    const _Float16* __restrict__ WihF,
                                    const _Float16* __restrict__ WihB,
                                    const float* __restrict__ bF,
                                    const float* __restrict__ bB,
                                    _Float16* __restrict__ PF,
                                    _Float16* __restrict__ PB) {
  __shared__ _Float16 ldsB[64 * E_];            // 64 KB
  const int dir  = blockIdx.z;
  const _Float16* W = dir ? WihB : WihF;
  const float*   bb = dir ? bB   : bF;
  _Float16*      P  = dir ? PB   : PF;

  const int lane = threadIdx.x & 31;
  const int wave = threadIdx.x >> 5;
  const long m0  = ((long)blockIdx.x * 8 + wave) * 16;
  const int  nb0 = blockIdx.y * 4;              // first of 4 N-tiles
  const _Float16* Wblk = W + (long)nb0 * 16 * E_;   // 64 contiguous rows

#if HAVE_TDM
  if (threadIdx.x < 32) {
    tdm_load_2d_f16(lds_byte_addr(&ldsB[0]), Wblk, E_, 64, E_);
    tdm_wait();
  }
#else
  for (int ci = threadIdx.x; ci < 64 * E_ / 8; ci += 256)
    *(v8h*)(&ldsB[(long)ci * 8]) = *(const v8h*)(Wblk + (long)ci * 8);
#endif
  __syncthreads();

  v8f acc0 = {}, acc1 = {}, acc2 = {}, acc3 = {};
  const _Float16* Abase = xe + m0 * E_;
  v16h a_cur = ld_frag16x32(Abase, E_, lane);
#pragma unroll 4
  for (int k0 = 0; k0 < E_; k0 += 32) {
    v16h a_nxt = a_cur;
    if (k0 + 32 < E_) {
      a_nxt = ld_frag16x32(Abase + k0 + 32, E_, lane);   // prefetch next chunk
      __builtin_prefetch(Abase + k0 + 64, 0, 1);
    }
    v16h b0 = ld_frag16x32(&ldsB[(0 * 16) * E_ + k0], E_, lane);
    v16h b1 = ld_frag16x32(&ldsB[(1 * 16) * E_ + k0], E_, lane);
    v16h b2 = ld_frag16x32(&ldsB[(2 * 16) * E_ + k0], E_, lane);
    v16h b3 = ld_frag16x32(&ldsB[(3 * 16) * E_ + k0], E_, lane);
    acc0 = __builtin_amdgcn_wmma_f32_16x16x32_f16(false, a_cur, false, b0, (short)0, acc0, false, false);
    acc1 = __builtin_amdgcn_wmma_f32_16x16x32_f16(false, a_cur, false, b1, (short)0, acc1, false, false);
    acc2 = __builtin_amdgcn_wmma_f32_16x16x32_f16(false, a_cur, false, b2, (short)0, acc2, false, false);
    acc3 = __builtin_amdgcn_wmma_f32_16x16x32_f16(false, a_cur, false, b3, (short)0, acc3, false, false);
    a_cur = a_nxt;
  }
  const int n   = lane & 15;
  const int mhi = (lane >> 4) << 3;
#pragma unroll
  for (int j = 0; j < 4; ++j) {
    v8f acc = (j == 0) ? acc0 : (j == 1) ? acc1 : (j == 2) ? acc2 : acc3;
    int col = (nb0 + j) * 16 + n;
    float bias = bb[col];
#pragma unroll
    for (int r = 0; r < 8; ++r) {
      long m = m0 + r + mhi;
      P[m * G4H + col] = (_Float16)(acc[r] + bias);
    }
  }
}

// ---------------------------------------------------------------------------
// 2) Persistent recurrent LSTM (one direction per launch).
//    32 blocks x 256 threads; block owns h-columns [j0, j0+16).
//    4x(16x512) W_hh gate slices TDM'd into LDS ONCE for all 512 steps.
//    Cell state c lives in WMMA accumulators; one grid barrier per step.
// ---------------------------------------------------------------------------
__device__ inline void grid_barrier(unsigned* cnt, unsigned nblk) {
  __threadfence();
  __syncthreads();
  if (threadIdx.x == 0) {
    unsigned ticket = atomicAdd(cnt, 1u);
    unsigned target = ticket - (ticket % nblk) + nblk;
    while (__hip_atomic_load(cnt, __ATOMIC_RELAXED, __HIP_MEMORY_SCOPE_AGENT) < target)
      __builtin_amdgcn_s_sleep(1);
  }
  __syncthreads();
}

__global__ void lstm_persistent_kernel(const _Float16* __restrict__ P,    // [BT,4H]
                                       const _Float16* __restrict__ Whh,  // [4H,H]
                                       _Float16* __restrict__ hs,         // [B,T,H]
                                       _Float16* __restrict__ hbuf0,
                                       _Float16* __restrict__ hbuf1,
                                       unsigned* ctr, int reverse) {
  __shared__ _Float16 ldsW[4 * 16 * H_];        // 64 KB
  const int tid   = threadIdx.x;
  const int lane  = tid & 31;
  const int wave  = tid >> 5;
  const int j0    = blockIdx.x * 16;
  const int bbase = wave * 16;

#if HAVE_TDM
  if (tid < 32) {
#pragma unroll
    for (int c = 0; c < 4; ++c)
      tdm_load_2d_f16(lds_byte_addr(&ldsW[(c * 16) * H_]),
                      Whh + (long)(c * 512 + j0) * H_, H_, 16, H_);
    tdm_wait();
  }
#else
  for (int ci = tid; ci < 4096; ci += 256) {
    int c  = ci >> 10;
    int nn = (ci >> 6) & 15;
    int ko = (ci & 63) << 3;
    *(v8h*)(&ldsW[((c * 16 + nn) << 9) + ko]) =
        *(const v8h*)(Whh + (long)(c * 512 + j0 + nn) * H_ + ko);
  }
#endif
  // zero both h double-buffers (1/32 stripe per block)
  for (int i = tid + blockIdx.x * 256; i < B_ * H_ / 8; i += 32 * 256) {
    *(v8h*)(hbuf0 + (long)i * 8) = (v8h){};
    *(v8h*)(hbuf1 + (long)i * 8) = (v8h){};
  }
  grid_barrier(ctr, 32);

  const int n   = lane & 15;
  const int mhi = (lane >> 4) << 3;
  v8f cst = {};

  for (int s = 0; s < T_; ++s) {
    const int t = reverse ? (T_ - 1 - s) : s;
    const _Float16* hcur = (s & 1) ? hbuf1 : hbuf0;
    _Float16*       hnxt = (s & 1) ? hbuf0 : hbuf1;
    const _Float16* Abase = hcur + (long)bbase * H_;

    v8f acc0 = {}, acc1 = {}, acc2 = {}, acc3 = {};
    v16h a_cur = ld_frag16x32(Abase, H_, lane);
#pragma unroll 4
    for (int k0 = 0; k0 < H_; k0 += 32) {
      v16h a_nxt = a_cur;
      if (k0 + 32 < H_) a_nxt = ld_frag16x32(Abase + k0 + 32, H_, lane);
      v16h b0 = ld_frag16x32(&ldsW[(0 << 13) + k0], H_, lane);
      v16h b1 = ld_frag16x32(&ldsW[(1 << 13) + k0], H_, lane);
      v16h b2 = ld_frag16x32(&ldsW[(2 << 13) + k0], H_, lane);
      v16h b3 = ld_frag16x32(&ldsW[(3 << 13) + k0], H_, lane);
      acc0 = __builtin_amdgcn_wmma_f32_16x16x32_f16(false, a_cur, false, b0, (short)0, acc0, false, false);
      acc1 = __builtin_amdgcn_wmma_f32_16x16x32_f16(false, a_cur, false, b1, (short)0, acc1, false, false);
      acc2 = __builtin_amdgcn_wmma_f32_16x16x32_f16(false, a_cur, false, b2, (short)0, acc2, false, false);
      acc3 = __builtin_amdgcn_wmma_f32_16x16x32_f16(false, a_cur, false, b3, (short)0, acc3, false, false);
      a_cur = a_nxt;
    }
#pragma unroll
    for (int r = 0; r < 8; ++r) {
      int  b    = bbase + r + mhi;
      long prow = ((long)b * T_ + t) * G4H;
      int  j    = j0 + n;
      float xi = acc0[r] + (float)P[prow + 0 * 512 + j];
      float xf = acc1[r] + (float)P[prow + 1 * 512 + j];
      float xg = acc2[r] + (float)P[prow + 2 * 512 + j];
      float xo = acc3[r] + (float)P[prow + 3 * 512 + j];
      float cc = sigf(xf) * cst[r] + sigf(xi) * tanhf(xg);
      cst[r] = cc;
      float hh = sigf(xo) * tanhf(cc);
      _Float16 h16 = (_Float16)hh;
      hnxt[(long)b * H_ + j] = h16;
      hs[((long)b * T_ + t) * H_ + j] = h16;
    }
    grid_barrier(ctr, 32);
  }
}

// ---------------------------------------------------------------------------
// 3) Output projection: em = [hf|hb] @ W_out^T + b_out   [BT, 32] f32
//    W_out (32x1024 f16 = 64KB) staged in LDS via TDM.
// ---------------------------------------------------------------------------
__global__ void proj_gemm_kernel(const _Float16* __restrict__ hf,
                                 const _Float16* __restrict__ hb,
                                 const _Float16* __restrict__ Wout,  // [32, 1024]
                                 const float* __restrict__ bout,
                                 float* __restrict__ em) {
  __shared__ _Float16 ldsW[L_ * 2 * H_];        // 64 KB
#if HAVE_TDM
  if (threadIdx.x < 32) {
    tdm_load_2d_f16(lds_byte_addr(&ldsW[0]), Wout, 2 * H_, L_, 2 * H_);
    tdm_wait();
  }
#else
  for (int ci = threadIdx.x; ci < L_ * 2 * H_ / 8; ci += 256)
    *(v8h*)(&ldsW[(long)ci * 8]) = *(const v8h*)(Wout + (long)ci * 8);
#endif
  __syncthreads();

  const int lane = threadIdx.x & 31;
  const int wave = threadIdx.x >> 5;
  const long m0  = ((long)blockIdx.x * 8 + wave) * 16;

  v8f acc0 = {}, acc1 = {};
  v16h a_cur = ld_frag16x32(hf + m0 * H_, H_, lane);
#pragma unroll 4
  for (int k0 = 0; k0 < 2 * H_; k0 += 32) {
    v16h a_nxt = a_cur;
    if (k0 + 32 < 2 * H_) {
      int kn = k0 + 32;
      const _Float16* nb = (kn < H_) ? (hf + m0 * H_ + kn) : (hb + m0 * H_ + (kn - H_));
      a_nxt = ld_frag16x32(nb, H_, lane);
    }
    v16h b0 = ld_frag16x32(&ldsW[0 * 16 * (2 * H_) + k0], 2 * H_, lane);
    v16h b1 = ld_frag16x32(&ldsW[1 * 16 * (2 * H_) + k0], 2 * H_, lane);
    acc0 = __builtin_amdgcn_wmma_f32_16x16x32_f16(false, a_cur, false, b0, (short)0, acc0, false, false);
    acc1 = __builtin_amdgcn_wmma_f32_16x16x32_f16(false, a_cur, false, b1, (short)0, acc1, false, false);
    a_cur = a_nxt;
  }
  const int n   = lane & 15;
  const int mhi = (lane >> 4) << 3;
#pragma unroll
  for (int j = 0; j < 2; ++j) {
    v8f acc = j ? acc1 : acc0;
    int col = j * 16 + n;
    float bias = bout[col];
#pragma unroll
    for (int r = 0; r < 8; ++r) {
      long m = m0 + r + mhi;
      em[m * L_ + col] = acc[r] + bias;
    }
  }
}

// ---------------------------------------------------------------------------
// 4) Viterbi decode: one wave32 per batch element, lane = label.
// ---------------------------------------------------------------------------
__global__ void viterbi_kernel(const float* __restrict__ em,
                               const float* __restrict__ trans,
                               const int* __restrict__ mask,
                               int* __restrict__ bp,
                               float* __restrict__ out) {
  __shared__ float tr[L_ * L_];
  const int b    = blockIdx.x;
  const int lane = threadIdx.x;
  for (int i = lane; i < L_ * L_; i += 32) tr[i] = trans[i];
  __syncthreads();

  float alpha = tr[BOS_TAG * L_ + lane] + em[((long)b * T_ + 0) * L_ + lane];
  for (int t = 1; t < T_; ++t) {
    float e  = em[((long)b * T_ + t) * L_ + lane];
    float mt = (float)mask[(long)b * T_ + t];
    float best = -3.4e38f; int arg = 0;
#pragma unroll 8
    for (int p = 0; p < L_; ++p) {
      float ap = __shfl(alpha, p, 32);
      float sc = ap + tr[p * L_ + lane] + e;
      if (sc > best) { best = sc; arg = p; }
    }
    bp[((long)b * T_ + t) * L_ + lane] = arg;
    alpha = mt * best + (1.0f - mt) * alpha;
  }
  float fin = alpha + tr[lane * L_ + EOS_TAG];
  float bv = fin; int bi = lane;
  for (int off = 16; off > 0; off >>= 1) {
    float ov = __shfl_xor(bv, off, 32);
    int   oi = __shfl_xor(bi, off, 32);
    if (ov > bv || (ov == bv && oi < bi)) { bv = ov; bi = oi; }
  }
  if (lane == 0) {
    out[b] = bv;
    float* path = out + B_ + (long)b * T_;
    int tag = bi;
    path[T_ - 1] = (float)tag;
    for (int t = T_ - 1; t >= 1; --t) {
      int prev = bp[((long)b * T_ + t) * L_ + tag];
      if (mask[(long)b * T_ + t] > 0) tag = prev;
      path[t - 1] = (float)tag;
    }
  }
}

// ---------------------------------------------------------------------------
// host-side launch
// ---------------------------------------------------------------------------
extern "C" void kernel_launch(void* const* d_in, const int* in_sizes, int n_in,
                              void* d_out, int out_size, void* d_ws, size_t ws_size,
                              hipStream_t stream) {
  const int*   x     = (const int*)d_in[0];
  const int*   mask  = (const int*)d_in[1];
  const float* emb   = (const float*)d_in[2];
  const float* WihF  = (const float*)d_in[3];
  const float* WhhF  = (const float*)d_in[4];
  const float* bF    = (const float*)d_in[5];
  const float* WihB  = (const float*)d_in[6];
  const float* WhhB  = (const float*)d_in[7];
  const float* bB    = (const float*)d_in[8];
  const float* Wout  = (const float*)d_in[9];
  const float* bout  = (const float*)d_in[10];
  const float* trans = (const float*)d_in[11];

  char* w = (char*)d_ws;
  size_t off = 0;
  auto nxt = [&](size_t bytes) -> char* {
    char* p = w + off;
    off = (off + bytes + 255) & ~(size_t)255;
    return p;
  };
  unsigned* ctr     = (unsigned*)nxt(256);
  _Float16* xe      = (_Float16*)nxt((size_t)BT * E_ * 2);
  _Float16* Wih16F  = (_Float16*)nxt((size_t)G4H * E_ * 2);
  _Float16* Wih16B  = (_Float16*)nxt((size_t)G4H * E_ * 2);
  _Float16* Whh16F  = (_Float16*)nxt((size_t)G4H * H_ * 2);
  _Float16* Whh16B  = (_Float16*)nxt((size_t)G4H * H_ * 2);
  _Float16* Wout16  = (_Float16*)nxt((size_t)L_ * 2 * H_ * 2);
  _Float16* PF      = (_Float16*)nxt((size_t)BT * G4H * 2);
  _Float16* PB      = (_Float16*)nxt((size_t)BT * G4H * 2);
  _Float16* hsF     = (_Float16*)nxt((size_t)BT * H_ * 2);
  _Float16* hsB     = (_Float16*)nxt((size_t)BT * H_ * 2);
  _Float16* hbF0    = (_Float16*)nxt((size_t)B_ * H_ * 2);
  _Float16* hbF1    = (_Float16*)nxt((size_t)B_ * H_ * 2);
  _Float16* hbB0    = (_Float16*)nxt((size_t)B_ * H_ * 2);
  _Float16* hbB1    = (_Float16*)nxt((size_t)B_ * H_ * 2);
  float*    em      = (float*)nxt((size_t)BT * L_ * 4);
  int*      bp      = (int*)nxt((size_t)BT * L_ * 4);

  zero_u32_kernel<<<1, 64, 0, stream>>>(ctr, 64);

  auto cvt = [&](const float* s, _Float16* d, long n) {
    cvt_f32_f16_kernel<<<(unsigned)((n + 255) / 256), 256, 0, stream>>>(s, d, n);
  };
  cvt(WihF, Wih16F, (long)G4H * E_);
  cvt(WihB, Wih16B, (long)G4H * E_);
  cvt(WhhF, Whh16F, (long)G4H * H_);
  cvt(WhhB, Whh16B, (long)G4H * H_);
  cvt(Wout, Wout16, (long)L_ * 2 * H_);

  embed_kernel<<<(unsigned)(((long)BT * E_ + 255) / 256), 256, 0, stream>>>(x, emb, xe);

  pregate_gemm_kernel<<<dim3(BT / 128, G4H / 64, 2), 256, 0, stream>>>(
      xe, Wih16F, Wih16B, bF, bB, PF, PB);

  lstm_persistent_kernel<<<32, 256, 0, stream>>>(PF, Whh16F, hsF, hbF0, hbF1, ctr + 0, 0);
  lstm_persistent_kernel<<<32, 256, 0, stream>>>(PB, Whh16B, hsB, hbB0, hbB1, ctr + 16, 1);

  proj_gemm_kernel<<<BT / 128, 256, 0, stream>>>(hsF, hsB, Wout16, bout, em);

  viterbi_kernel<<<B_, 32, 0, stream>>>(em, trans, mask, bp, (float*)d_out);
}
